// VisionAgentAttention_60559038873833
// MI455X (gfx1250) — matrix operand order repaired and checked by reading, not compiled, not run on hardware
//
#include <hip/hip_runtime.h>
#include <hip/hip_bf16.h>

typedef __attribute__((ext_vector_type(16))) _Float16 v16h;
typedef __attribute__((ext_vector_type(8)))  _Float16 v8h;
typedef __attribute__((ext_vector_type(8)))  float    v8f;

#define DEV static __device__ __forceinline__

// exp(x * 1/sqrt(32)) == exp2(x * (log2e/sqrt(32)))
#define EXP2_SCALE 0.2550332785f

DEV v8f wmma32(v16h a, v16h b, v8f c) {
    // D = A(16x32 f16) * B(32x16 f16) + C(16x16 f32)
    return __builtin_amdgcn_wmma_f32_16x16x32_f16(
        false, a, false, b, (short)0, c, false, false);
}

// A-fragment (16x32, f16): lane holds row M=L%16; halves e -> K = (hi?16:0)+e.
// `row` points at 32 contiguous halves.
DEV v16h load_afrag(const _Float16* row, int hi) {
    v8h r0 = *(const v8h*)(row + (hi ? 16 : 0));
    v8h r1 = *(const v8h*)(row + (hi ? 24 : 8));
    v16h a;
#pragma unroll
    for (int e = 0; e < 8; ++e) { a[e] = r0[e]; a[8 + e] = r1[e]; }
    return a;
}

// B-fragment (32x16, f16): lane holds col N=L%16; halves e -> K = (hi?16:0)+e.
// `p` points at the 16 contiguous halves for this lane.
DEV v16h load_bfrag(const _Float16* p) {
    v8h r0 = *(const v8h*)p;
    v8h r1 = *(const v8h*)(p + 8);
    v16h b;
#pragma unroll
    for (int e = 0; e < 8; ++e) { b[e] = r0[e]; b[8 + e] = r1[e]; }
    return b;
}

DEV void load8f(const float* p, v16h& dst, int off) {
    float4 f0 = *(const float4*)p;
    float4 f1 = *(const float4*)(p + 4);
    dst[off + 0] = (_Float16)f0.x; dst[off + 1] = (_Float16)f0.y;
    dst[off + 2] = (_Float16)f0.z; dst[off + 3] = (_Float16)f0.w;
    dst[off + 4] = (_Float16)f1.x; dst[off + 5] = (_Float16)f1.y;
    dst[off + 6] = (_Float16)f1.z; dst[off + 7] = (_Float16)f1.w;
}

// ---------------------------------------------------------------------------
// f32 -> f16 weight pre-conversion (once per weight tensor, 768*256 elements)
// ---------------------------------------------------------------------------
__global__ __launch_bounds__(256) void cvt_kernel(
    const float* __restrict__ src, _Float16* __restrict__ dst, int n)
{
    int i = blockIdx.x * blockDim.x + threadIdx.x;
    if (i < n) dst[i] = (_Float16)src[i];
}

// ---------------------------------------------------------------------------
// Patch embed: Conv(16x16 stride 16) as GEMM  [Npatch x 768] @ [768 x 256]
// + bias + sinusoidal pos-enc. One wave owns a 16-patch x 256-channel strip:
// 16 f32 accumulators, A-fragment (im2col, f32->f16 inline) loaded once per
// K-chunk and reused by 16 independent WMMAs. G/IMGW compile-time so all
// im2col offsets fold into load immediates (lane-half offset is just +IMGW*hi).
// Output f16: (b,h,n,32) or transposed (b,h,32,n) for V.
// ---------------------------------------------------------------------------
template<int G, int IMGW, int TRANSPOSED>
__global__ __launch_bounds__(256) void pe_kernel(
    const float* __restrict__ x, const _Float16* __restrict__ wh,
    const float* __restrict__ bias, _Float16* __restrict__ out)
{
    constexpr int Np = G * G;
    constexpr int mTiles = Np / 16;
    const int wave = (blockIdx.x * blockDim.x + threadIdx.x) >> 5;
    const int lane = threadIdx.x & 31;
    const int ln = lane & 15, hi = lane >> 4;
    const int mTile = wave % mTiles;
    const int b = wave / mTiles;

    const int m = mTile * 16 + ln;            // patch row for A-frag
    const int gy = m / G, gx = m % G;

    v8f acc[16];
#pragma unroll
    for (int nt = 0; nt < 16; ++nt) acc[nt] = v8f{};

    // base of this lane's patch; +IMGW*hi selects the patch row (i) for the lane half
    const float* xb = x + ((size_t)(b * 3) * IMGW + (size_t)gy * 16) * IMGW
                        + gx * 16 + hi * IMGW;
    const _Float16* wB = wh + (size_t)ln * 768 + hi * 16;

#pragma unroll
    for (int kc = 0; kc < 24; ++kc) {
        // im2col offsets: k = c*256 + i*16 + j -> (c*IMGW + i)*IMGW + j
        const int k0 = kc * 32, k1 = kc * 32 + 16;
        const int off0 = ((k0 >> 8) * IMGW + ((k0 >> 4) & 15)) * IMGW;
        const int off1 = ((k1 >> 8) * IMGW + ((k1 >> 4) & 15)) * IMGW;
        v16h a;
        load8f(xb + off0, a, 0);
        load8f(xb + off0 + 8, a, 8);
#pragma unroll
        for (int nt = 0; nt < 16; ++nt) {
            v16h bf = load_bfrag(wB + nt * (16 * 768) + kc * 32);
            acc[nt] = wmma32(a, bf, acc[nt]);
        }
    }

    // epilogue: bias + positional encoding, store f16
    const int mb = mTile * 16 + (hi ? 8 : 0);
#pragma unroll
    for (int nt = 0; nt < 16; ++nt) {
        const int d = nt * 16 + ln;
        const float bv = bias[d];
        const float freq = __expf(-(float)(d & ~1) * 0.0359778921f); // ln(1e4)/256
        const int h = d >> 5, hd = d & 31;
        if (!TRANSPOSED) {
            _Float16* op = out + ((size_t)(b * 8 + h) * Np + mb) * 32 + hd;
#pragma unroll
            for (int v = 0; v < 8; ++v) {
                float ang = (float)(mb + v) * freq;
                float pe = (d & 1) ? __cosf(ang) : __sinf(ang);
                op[v * 32] = (_Float16)(acc[nt][v] + bv + pe);
            }
        } else {
            v8h pk;
#pragma unroll
            for (int v = 0; v < 8; ++v) {
                float ang = (float)(mb + v) * freq;
                float pe = (d & 1) ? __cosf(ang) : __sinf(ang);
                pk[v] = (_Float16)(acc[nt][v] + bv + pe);
            }
            *(v8h*)(out + ((size_t)(b * 8 + h) * 32 + hd) * Np + mb) = pk;
        }
    }
}

// ---------------------------------------------------------------------------
// K1: per (b,h): Ks = softmax_n(ah @ kh^T * s) ; KV^T = (Ks @ vh)^T  (f16)
// One 16-agent-row tile per wave; stream Nk=1024 in 32-wide chunks:
//   logits (2 WMMA, K=32=HD) -> exp2 -> LDS transpose -> KV accum (2 WMMA).
// Bases hoisted per-lane, chunk loop fully unrolled -> immediate offsets.
// Row-sums reduced with shfl_xor inside each 16-lane half.
// ---------------------------------------------------------------------------
__global__ __launch_bounds__(256) void kv_kernel(
    const _Float16* __restrict__ ah, const _Float16* __restrict__ kh,
    const _Float16* __restrict__ vht, _Float16* __restrict__ kvt)
{
    __shared__ __align__(16) _Float16 lds[8 * 512];
    const int wslot = threadIdx.x >> 5;
    const int wave = (blockIdx.x * blockDim.x + threadIdx.x) >> 5;
    const int lane = threadIdx.x & 31;
    const int ln = lane & 15, hi = lane >> 4;
    const int bh = wave >> 4;          // 128 (b,h) pairs
    const int rt = wave & 15;          // agent row tile (256/16)
    _Float16* tile = lds + wslot * 512;

    v16h aF = load_afrag(ah + ((size_t)bh * 256 + rt * 16 + ln) * 32, hi);

    // per-lane base pointers; all chunk loads become base + immediate
    const _Float16* kbase = kh + ((size_t)bh * 1024 + ln) * 32 + (hi ? 16 : 0);
    const _Float16* vbase = vht + ((size_t)bh * 32 + ln) * 1024 + (hi ? 16 : 0);
    _Float16* tst = tile + (hi ? 8 : 0) * 32 + ln;   // store base for exp tile

    v8f kv0 = {}, kv1 = {};
    float rs[8];
#pragma unroll
    for (int v = 0; v < 8; ++v) rs[v] = 0.f;

#pragma unroll
    for (int nc = 0; nc < 32; ++nc) {
#pragma unroll
        for (int s = 0; s < 2; ++s) {
            v16h bF = load_bfrag(kbase + (nc * 32 + s * 16) * 32);
            v8f c = {};
            c = wmma32(aF, bF, c);
#pragma unroll
            for (int v = 0; v < 8; ++v) {
                float e = __builtin_amdgcn_exp2f(c[v] * EXP2_SCALE);
                rs[v] += e;
                tst[v * 32 + s * 16] = (_Float16)e;
            }
        }
        __syncthreads();
        v16h eF = load_afrag(tile + (size_t)ln * 32, hi);  // exp tile as A (16x32)
        v16h b0 = load_bfrag(vbase + nc * 32);              // hd cols 0..15
        v16h b1 = load_bfrag(vbase + 16 * 1024 + nc * 32);  // hd cols 16..31
        kv0 = wmma32(eF, b0, kv0);
        kv1 = wmma32(eF, b1, kv1);
        __syncthreads();
    }

#pragma unroll
    for (int v = 0; v < 8; ++v) {
        float s = rs[v];
        s += __shfl_xor(s, 1); s += __shfl_xor(s, 2);
        s += __shfl_xor(s, 4); s += __shfl_xor(s, 8);
        rs[v] = __builtin_amdgcn_rcpf(s);
    }
    const int mb = rt * 16 + (hi ? 8 : 0);
    v8h p0, p1;
#pragma unroll
    for (int v = 0; v < 8; ++v) {
        p0[v] = (_Float16)(kv0[v] * rs[v]);
        p1[v] = (_Float16)(kv1[v] * rs[v]);
    }
    *(v8h*)(kvt + ((size_t)bh * 32 + ln) * 256 + mb) = p0;
    *(v8h*)(kvt + ((size_t)bh * 32 + 16 + ln) * 256 + mb) = p1;
}

// ---------------------------------------------------------------------------
// K2: per (b,h): Qs = softmax_m(qh @ ah^T * s) ; out = Qs @ KV, re-tiled into
// the final (B,512,512) spatial layout. One 16-q-row tile per wave,
// Na=256 streamed in 8 fully-unrolled chunks of 32.
// ---------------------------------------------------------------------------
__global__ __launch_bounds__(256) void out_kernel(
    const _Float16* __restrict__ qh, const _Float16* __restrict__ ah,
    const _Float16* __restrict__ kvt, float* __restrict__ out)
{
    __shared__ __align__(16) _Float16 lds[8 * 512];
    const int wslot = threadIdx.x >> 5;
    const int wave = (blockIdx.x * blockDim.x + threadIdx.x) >> 5;
    const int lane = threadIdx.x & 31;
    const int ln = lane & 15, hi = lane >> 4;
    const int bh = wave >> 6;          // 128 (b,h) pairs
    const int rt = wave & 63;          // q row tile (1024/16)
    _Float16* tile = lds + wslot * 512;

    v16h aF = load_afrag(qh + ((size_t)bh * 1024 + rt * 16 + ln) * 32, hi);

    const _Float16* abase = ah + ((size_t)bh * 256 + ln) * 32 + (hi ? 16 : 0);
    const _Float16* kvbase = kvt + ((size_t)bh * 32 + ln) * 256 + (hi ? 16 : 0);
    _Float16* tst = tile + (hi ? 8 : 0) * 32 + ln;

    v8f o0 = {}, o1 = {};
    float rs[8];
#pragma unroll
    for (int v = 0; v < 8; ++v) rs[v] = 0.f;

#pragma unroll
    for (int ac = 0; ac < 8; ++ac) {
#pragma unroll
        for (int s = 0; s < 2; ++s) {
            v16h bF = load_bfrag(abase + (ac * 32 + s * 16) * 32);
            v8f c = {};
            c = wmma32(aF, bF, c);
#pragma unroll
            for (int v = 0; v < 8; ++v) {
                float e = __builtin_amdgcn_exp2f(c[v] * EXP2_SCALE);
                rs[v] += e;
                tst[v * 32 + s * 16] = (_Float16)e;
            }
        }
        __syncthreads();
        v16h eF = load_afrag(tile + (size_t)ln * 32, hi);
        v16h b0 = load_bfrag(kvbase + ac * 32);
        v16h b1 = load_bfrag(kvbase + 16 * 256 + ac * 32);
        o0 = wmma32(eF, b0, o0);
        o1 = wmma32(eF, b1, o1);
        __syncthreads();
    }

#pragma unroll
    for (int v = 0; v < 8; ++v) {
        float s = rs[v];
        s += __shfl_xor(s, 1); s += __shfl_xor(s, 2);
        s += __shfl_xor(s, 4); s += __shfl_xor(s, 8);
        rs[v] = __builtin_amdgcn_rcpf(s);
    }

    // spatial re-tile: d = h*32+hd, pp=d>>4, pq=d&15; y=gy*16+pp, x=gx*16+pq
    // n-run of 8 (v=0..7) stays inside one gy row -> stores are base + v*16.
    const int b = bh >> 3, h = bh & 7;
    const int n0 = rt * 16 + (hi ? 8 : 0);
    const int gy = n0 >> 5, gx0 = n0 & 31;
#pragma unroll
    for (int ct = 0; ct < 2; ++ct) {
        int hd = ct * 16 + ln;
        int dd = h * 32 + hd;
        int pp = dd >> 4, pq = dd & 15;
        float* op = out + (size_t)b * 262144 + (size_t)(gy * 16 + pp) * 512
                        + gx0 * 16 + pq;
#pragma unroll
        for (int v = 0; v < 8; ++v)
            op[v * 16] = (ct ? o1[v] : o0[v]) * rs[v];
    }
}

// ---------------------------------------------------------------------------
extern "C" void kernel_launch(void* const* d_in, const int* in_sizes, int n_in,
                              void* d_out, int out_size, void* d_ws, size_t ws_size,
                              hipStream_t stream) {
    (void)in_sizes; (void)n_in; (void)out_size; (void)ws_size;
    const float* q  = (const float*)d_in[0];
    const float* k  = (const float*)d_in[1];
    const float* v  = (const float*)d_in[2];
    const float* a  = (const float*)d_in[3];
    const float* wq = (const float*)d_in[4];  const float* bq = (const float*)d_in[5];
    const float* wk = (const float*)d_in[6];  const float* bk = (const float*)d_in[7];
    const float* wv = (const float*)d_in[8];  const float* bv = (const float*)d_in[9];
    const float* wa = (const float*)d_in[10]; const float* ba = (const float*)d_in[11];

    // workspace: f16 intermediates, (b,h)-major
    const size_t QH = (size_t)128 * 1024 * 32;  // qh/kh/vht: 8 MB each
    const size_t AH = (size_t)128 * 256 * 32;   // ah/kvt:    2 MB each
    const size_t WH = (size_t)768 * 256;        // f16 weights: 384 KB each
    _Float16* base = (_Float16*)d_ws;
    _Float16* qh  = base;
    _Float16* kh  = qh + QH;
    _Float16* vht = kh + QH;
    _Float16* ahh = vht + QH;
    _Float16* kvt = ahh + AH;
    _Float16* whq = kvt + AH;
    _Float16* whk = whq + WH;
    _Float16* whv = whk + WH;
    _Float16* wha = whv + WH;

    // weight f32 -> f16 (once)
    cvt_kernel<<<768, 256, 0, stream>>>(wq, whq, (int)WH);
    cvt_kernel<<<768, 256, 0, stream>>>(wk, whk, (int)WH);
    cvt_kernel<<<768, 256, 0, stream>>>(wv, whv, (int)WH);
    cvt_kernel<<<768, 256, 0, stream>>>(wa, wha, (int)WH);

    // patch embeds: one wave per 16-patch strip -> waves = B * (Np/16)
    pe_kernel<32, 512, 0><<<128, 256, 0, stream>>>(q, whq, bq, qh);
    pe_kernel<32, 512, 0><<<128, 256, 0, stream>>>(k, whk, bk, kh);
    pe_kernel<32, 512, 1><<<128, 256, 0, stream>>>(v, whv, bv, vht);
    pe_kernel<16, 256, 0><<< 32, 256, 0, stream>>>(a, wha, ba, ahh);

    // attention
    kv_kernel<<< 256, 256, 0, stream>>>(ahh, kh, vht, kvt);   // 128*16 waves
    out_kernel<<<1024, 256, 0, stream>>>(qh, ahh, kvt, (float*)d_out); // 128*64 waves
}
